// PhrasalLexemeAttention_21998822490644
// MI455X (gfx1250) — compile-verified, matrix operand order checked by reading, not run on hardware
//
#include <hip/hip_runtime.h>
#include <hip/hip_bf16.h>

typedef __attribute__((ext_vector_type(16))) __bf16 v16bf;
typedef __attribute__((ext_vector_type(8)))  float  v8f;
typedef __attribute__((ext_vector_type(4)))  float  f4;

#define BS     4
#define SEQ    1024
#define DMODEL 512
#define NHEAD  8
#define DQ     64
#define NMAT   (NHEAD * DQ)     // 512
#define MROWS  (BS * SEQ)       // 4096
#define NEG_INF_F (-1e9f)
#define EPS_F     (1e-9f)
#define SQRT_EPS_F 3.16227766e-05f
#define TOT ((size_t)BS * NHEAD * SEQ * SEQ)   // 33,554,432 per tensor

// ---------------------------------------------------------------------------
// Kernel 1: Out = X @ W^T + bias  via v_wmma_f32_16x16x32_bf16
// X: (MROWS, DMODEL) f32, W: (NMAT, DMODEL) f32, Out: (MROWS, NMAT) f32
// One wave computes a 16(M) x 64(N) strip; K loop in steps of 32.
// Fragment layouts per CDNA5 ISA 7.12.2 (16-bit A 16x32 / B 32x16):
//   lane = M (or N) index (lane&15), half = lane>>4
//   per-lane element e maps to K = e + 8*half + (e>=8 ? 8 : 0)
// ---------------------------------------------------------------------------
__global__ __launch_bounds__(256)
void proj_wmma(const float* __restrict__ X, const float* __restrict__ W,
               const float* __restrict__ bias, float* __restrict__ Out)
{
    const int wave = (blockIdx.x * blockDim.x + threadIdx.x) >> 5;
    const int lane = threadIdx.x & 31;
    const int m0   = (wave >> 3) * 16;   // 256 row strips
    const int n0   = (wave & 7) * 64;    // 8 col strips
    const int row  = lane & 15;
    const int half = lane >> 4;

    v8f acc[4] = {};
    const float* xrow = X + (size_t)(m0 + row) * DMODEL;

    for (int kk = 0; kk < DMODEL; kk += 32) {
        v16bf a;
        {
            const float* p = xrow + kk + half * 8;
            #pragma unroll
            for (int e = 0; e < 8; ++e) a[e]     = (__bf16)p[e];
            #pragma unroll
            for (int e = 0; e < 8; ++e) a[e + 8] = (__bf16)p[16 + e];
        }
        #pragma unroll
        for (int t = 0; t < 4; ++t) {
            const float* wrow =
                W + (size_t)(n0 + t * 16 + row) * DMODEL + kk + half * 8;
            v16bf b;
            #pragma unroll
            for (int e = 0; e < 8; ++e) b[e]     = (__bf16)wrow[e];
            #pragma unroll
            for (int e = 0; e < 8; ++e) b[e + 8] = (__bf16)wrow[16 + e];
            acc[t] = __builtin_amdgcn_wmma_f32_16x16x32_bf16(
                false, a, false, b, (short)0, acc[t], false, false);
        }
    }

    // D layout: element r, lane -> M = m0 + r + 8*half, N = n0 + t*16 + (lane&15)
    #pragma unroll
    for (int t = 0; t < 4; ++t) {
        const int n = n0 + t * 16 + row;
        const float bv = bias[n];
        #pragma unroll
        for (int r = 0; r < 8; ++r) {
            const int m = m0 + r + half * 8;
            Out[(size_t)m * NMAT + n] = acc[t][r] + bv;
        }
    }
}

// ---------------------------------------------------------------------------
// Kernel 2: per (b,h): neighbor scores -> 2-entry softmax -> band -> prefix C
// ---------------------------------------------------------------------------
__global__ __launch_bounds__(256)
void band_scan(const float* __restrict__ Q, const float* __restrict__ K,
               const int* __restrict__ amask,
               float* __restrict__ band, float* __restrict__ Cpre)
{
    const int b  = blockIdx.x >> 3;
    const int h  = blockIdx.x & 7;
    const int bh = blockIdx.x;

    __shared__ float p_next[SEQ];
    __shared__ float p_prev[SEQ];
    __shared__ float bnd[SEQ];

    const size_t off = (size_t)b * SEQ * NMAT + (size_t)h * DQ;

    for (int i = threadIdx.x; i < SEQ; i += blockDim.x) {
        const float* q = Q + off + (size_t)i * NMAT;
        float sfwd = NEG_INF_F, sbwd = NEG_INF_F;
        if (i + 1 < SEQ && amask[b * SEQ + i + 1]) {
            const float* kv = K + off + (size_t)(i + 1) * NMAT;
            float acc = 0.f;
            #pragma unroll
            for (int d = 0; d < DQ; d += 4) {
                f4 qa = *(const f4*)(q + d);
                f4 ka = *(const f4*)(kv + d);
                acc += qa.x * ka.x + qa.y * ka.y + qa.z * ka.z + qa.w * ka.w;
            }
            sfwd = acc * (1.0f / DMODEL);
        }
        if (i - 1 >= 0 && amask[b * SEQ + i - 1]) {
            const float* kv = K + off + (size_t)(i - 1) * NMAT;
            float acc = 0.f;
            #pragma unroll
            for (int d = 0; d < DQ; d += 4) {
                f4 qa = *(const f4*)(q + d);
                f4 ka = *(const f4*)(kv + d);
                acc += qa.x * ka.x + qa.y * ka.y + qa.z * ka.z + qa.w * ka.w;
            }
            sbwd = acc * (1.0f / DMODEL);
        }
        const float m  = fmaxf(sfwd, sbwd);
        const float ea = expf(sbwd - m);
        const float eb = expf(sfwd - m);
        const float inv = 1.0f / (ea + eb);
        p_prev[i] = ea * inv;
        p_next[i] = eb * inv;
    }
    __syncthreads();

    for (int i = threadIdx.x; i < SEQ; i += blockDim.x) {
        const float v = (i < SEQ - 1)
                      ? sqrtf(p_next[i] * p_prev[i + 1] + EPS_F) : 0.0f;
        bnd[i] = v;
        band[(size_t)bh * SEQ + i] = v;
    }
    __syncthreads();

    if (threadIdx.x == 0) {   // 1023-step serial scan: negligible
        float c = 0.f;
        Cpre[(size_t)bh * SEQ] = 0.f;
        for (int i = 1; i < SEQ; ++i) {
            c += logf(bnd[i - 1] + EPS_F);
            Cpre[(size_t)bh * SEQ + i] = c;
        }
    }
}

// ---------------------------------------------------------------------------
// Kernel 3: streaming fill of both outputs (268 MB total; the roofline term).
// attn(i,k)    = i==k ? sqrt(EPS) : exp(C(max)-C(min)) + EPS
// phrasal(i,k) = |i-k|==1 ? band(min) : sqrt(EPS)
// 4 elems/thread, b128 nontemporal stores.
// ---------------------------------------------------------------------------
__global__ __launch_bounds__(256)
void fill_out(const float* __restrict__ band, const float* __restrict__ Cpre,
              float* __restrict__ out)
{
    const size_t tid  = (size_t)blockIdx.x * blockDim.x + threadIdx.x;
    const size_t base = tid * 4;
    const int bh  = (int)(base >> 20);            // SEQ*SEQ = 2^20
    const int rem = (int)(base & 0xFFFFFu);
    const int i   = rem >> 10;
    const int k4  = rem & 1023;

    const float* C = Cpre + (size_t)bh * SEQ;
    const float* B = band + (size_t)bh * SEQ;
    const float Ci = C[i];

    float av[4], pv[4];
    #pragma unroll
    for (int j = 0; j < 4; ++j) {
        const int k = k4 + j;
        const int d = k - i;
        if (d == 0) {
            av[j] = SQRT_EPS_F;
            pv[j] = SQRT_EPS_F;
        } else {
            const float Ck = C[k];
            const float hi = (d > 0) ? Ck : Ci;
            const float lo = (d > 0) ? Ci : Ck;
            av[j] = expf(hi - lo) + EPS_F;
            pv[j] = (d == 1) ? B[i] : ((d == -1) ? B[k] : SQRT_EPS_F);
        }
    }
    f4 oa = { av[0], av[1], av[2], av[3] };
    f4 op = { pv[0], pv[1], pv[2], pv[3] };
    __builtin_nontemporal_store(oa, (f4*)(out + base));
    __builtin_nontemporal_store(op, (f4*)(out + TOT + base));
}

// ---------------------------------------------------------------------------
extern "C" void kernel_launch(void* const* d_in, const int* in_sizes, int n_in,
                              void* d_out, int out_size, void* d_ws, size_t ws_size,
                              hipStream_t stream)
{
    const float* context = (const float*)d_in[0];
    const int*   amask   = (const int*)  d_in[1];
    const float* Wq      = (const float*)d_in[2];
    const float* bq      = (const float*)d_in[3];
    const float* Wk      = (const float*)d_in[4];
    const float* bk      = (const float*)d_in[5];
    float* out = (float*)d_out;

    float* Qws  = (float*)d_ws;                       // 4096*512
    float* Kws  = Qws + (size_t)MROWS * NMAT;         // 4096*512
    float* band = Kws + (size_t)MROWS * NMAT;         // 32*1024
    float* Cpre = band + (size_t)BS * NHEAD * SEQ;    // 32*1024

    dim3 blk(256);
    // (4096/16)*(512/64) = 2048 waves, 8 waves/block -> 256 blocks
    proj_wmma<<<256, blk, 0, stream>>>(context, Wq, bq, Qws);
    proj_wmma<<<256, blk, 0, stream>>>(context, Wk, bk, Kws);
    band_scan<<<BS * NHEAD, blk, 0, stream>>>(Qws, Kws, amask, band, Cpre);
    // TOT/4 threads / 256 = 32768 blocks (exact)
    fill_out<<<(unsigned)(TOT / 4 / 256), blk, 0, stream>>>(band, Cpre, out);
}
